// Polarnet_60593398612299
// MI455X (gfx1250) — matrix operand order repaired and checked by reading, not compiled, not run on hardware
//
#include <hip/hip_runtime.h>

typedef __attribute__((ext_vector_type(16))) _Float16 v16h;
typedef __attribute__((ext_vector_type(8)))  float    v8f;

#define N_PTS  300000
#define BATCH  4
#define GX     480
#define GY     360
#define MAX_PT 64
#define EPSBN  1e-5f

#define NVOX   (BATCH*GX*GY)          // 691200
#define POOLN  ((size_t)NVOX*64)      // 44,236,800 floats

// workspace layout (bytes)
#define OFF_W    ((size_t)0)          // 59392 f16 = 118,784 B (fragment-ordered weights)
#define OFF_BIAS ((size_t)131072)     // 512 f32
#define OFF_VOX  ((size_t)135168)     // N_PTS int (vox id or -1 if dropped)
#define OFF_CNT  ((size_t)1376256)    // NVOX int
#define OFF_POOL ((size_t)4194304)    // POOLN f32 in [B,C,GX,GY] layout

// fragment bases (in 512-half fragment units): L1=0(4), L2=4(16), L3=20(64), L4=84(32)

__device__ __forceinline__ void store_frag(_Float16* wf, int baseFrag, int NT,
                                           int k, int n, float val) {
    int kt = k >> 5, kk = k & 31, nt = n >> 4, nn = n & 15;
    int lane = (kk & 16) | nn;      // lanes 16-31 hold K=16..31 of the 32-K slab
    int h = kk & 15;                // element within lane's 16 halves
    wf[((size_t)(baseFrag + kt * NT + nt) * 32 + lane) * 16 + h] = (_Float16)val;
}

// ---------------- prep: fold BN into weights/biases, emit f16 B-fragments ---------------
__global__ void prep_kernel(
    const float* g0, const float* b0, const float* m0, const float* v0,
    const float* g1, const float* b1, const float* m1, const float* v1,
    const float* g2, const float* b2, const float* m2, const float* v2,
    const float* g3, const float* b3, const float* m3, const float* v3,
    const float* w1, const float* bb1, const float* w2, const float* bb2,
    const float* w3, const float* bb3, const float* w4, const float* bb4,
    _Float16* wf, float* bias)
{
    int gid = blockIdx.x * blockDim.x + threadIdx.x;
    int gsz = gridDim.x * blockDim.x;

    for (int n = gid; n < 64; n += gsz) {
        float s1 = g1[n] * rsqrtf(v1[n] + EPSBN);
        float t1 = b1[n] - m1[n] * s1;
        float acc = bb1[n];
        for (int k = 0; k < 9; ++k) {
            float s0 = g0[k] * rsqrtf(v0[k] + EPSBN);
            float t0 = b0[k] - m0[k] * s0;
            acc += t0 * w1[k * 64 + n];
        }
        bias[n] = acc * s1 + t1;
    }
    for (int n = gid; n < 128; n += gsz) {
        float s = g2[n] * rsqrtf(v2[n] + EPSBN);
        bias[64 + n] = bb2[n] * s + (b2[n] - m2[n] * s);
    }
    for (int n = gid; n < 256; n += gsz) {
        float s = g3[n] * rsqrtf(v3[n] + EPSBN);
        bias[192 + n] = bb3[n] * s + (b3[n] - m3[n] * s);
    }
    for (int n = gid; n < 64; n += gsz) bias[448 + n] = bb4[n];

    for (int i = gid; i < 32 * 64; i += gsz) {          // L1: K pad 9->32, N=64
        int k = i >> 6, n = i & 63;
        float val = 0.f;
        if (k < 9) {
            float s0 = g0[k] * rsqrtf(v0[k] + EPSBN);
            float s1 = g1[n] * rsqrtf(v1[n] + EPSBN);
            val = s0 * w1[k * 64 + n] * s1;
        }
        store_frag(wf, 0, 4, k, n, val);
    }
    for (int i = gid; i < 64 * 128; i += gsz) {         // L2
        int k = i >> 7, n = i & 127;
        float s = g2[n] * rsqrtf(v2[n] + EPSBN);
        store_frag(wf, 4, 8, k, n, w2[k * 128 + n] * s);
    }
    for (int i = gid; i < 128 * 256; i += gsz) {        // L3
        int k = i >> 8, n = i & 255;
        float s = g3[n] * rsqrtf(v3[n] + EPSBN);
        store_frag(wf, 20, 16, k, n, w3[k * 256 + n] * s);
    }
    for (int i = gid; i < 256 * 64; i += gsz) {         // L4
        int k = i >> 6, n = i & 63;
        store_frag(wf, 84, 4, k, n, w4[k * 64 + n]);
    }
}

// ---------------- init: pooled = -inf, counters = 0 ----------------
__global__ void init_kernel(unsigned* pooled, int* cnt) {
    size_t i = (size_t)blockIdx.x * blockDim.x + threadIdx.x;
    size_t stride = (size_t)gridDim.x * blockDim.x;
    for (size_t j = i; j < POOLN; j += stride) pooled[j] = 0xFF800000u;  // -inf
    for (size_t j = i; j < (size_t)NVOX; j += stride) cnt[j] = 0;
}

// ---------------- rank: voxel id + keep (first MAX_PT arrivals) ----------------
__global__ void rank_kernel(const int* bi, const int* gx, const int* gy,
                            int* cnt, int* voxkeep) {
    int p = blockIdx.x * blockDim.x + threadIdx.x;
    if (p >= N_PTS) return;
    int v = (bi[p] * GX + gx[p]) * GY + gy[p];
    int r = atomicAdd(&cnt[v], 1);
    voxkeep[p] = (r < MAX_PT) ? v : -1;
}

// ---------------- WMMA MLP ----------------
__device__ __forceinline__ void atomicMaxFloat(float* addr, float val) {
    if (val >= 0.f) atomicMax((int*)addr, __float_as_int(val));
    else            atomicMin((unsigned int*)addr, __float_as_uint(val));
}

template<int KT>
__device__ __forceinline__ void load_a(const _Float16 (*A)[264], int lane, v16h* a) {
    int m  = lane & 15;
    int kb = (lane & 16) ? 16 : 0;  // lanes 16-31 hold K 16..31 (contiguous)
#pragma unroll
    for (int kt = 0; kt < KT; ++kt) {
        union { v16h v; unsigned u[8]; } U;
#pragma unroll
        for (int p = 0; p < 8; ++p) {
            int k0 = kt * 32 + kb + (p << 1);
            U.u[p] = *(const unsigned*)&A[m][k0];   // pair of halves
        }
        a[kt] = U.v;
    }
}

template<int KT, int NT>
__device__ __forceinline__ void mlp_layer(_Float16 (*A)[264], const _Float16* wf,
                                          int baseFrag, const float* bias, int biasOff,
                                          int lane) {
    v16h a[KT];
    load_a<KT>(A, lane, a);
    int nn  = lane & 15;
    int hi8 = (lane & 16) ? 8 : 0;
#pragma unroll
    for (int nt = 0; nt < NT; ++nt) {
        v8f acc = {};
#pragma unroll
        for (int kt = 0; kt < KT; ++kt) {
            v16h b = *(const v16h*)(wf + ((size_t)(baseFrag + kt * NT + nt) * 32 + lane) * 16);
            acc = __builtin_amdgcn_wmma_f32_16x16x32_f16(false, a[kt], false, b,
                                                         (short)0, acc, false, false);
        }
        float bv = bias[biasOff + nt * 16 + nn];
#pragma unroll
        for (int r = 0; r < 8; ++r) {
            float v = acc[r] + bv;
            v = v > 0.f ? v : 0.f;                       // ReLU (BN already folded)
            A[r + hi8][nt * 16 + nn] = (_Float16)v;
        }
    }
}

__launch_bounds__(128)
__global__ void mlp_kernel(const float* __restrict__ pt, const int* __restrict__ voxkeep,
                           const _Float16* __restrict__ wf, const float* __restrict__ bias,
                           float* __restrict__ pooled, int ntiles) {
    __shared__ _Float16 act[4][16][264];   // per-wave 16x(<=256) activation tile, padded
    int wave = threadIdx.x >> 5;
    int lane = threadIdx.x & 31;
    int tile = blockIdx.x * 4 + wave;
    if (tile >= ntiles) return;            // no block barriers used -> safe early exit
    int p0 = tile * 16;

    _Float16 (*A)[264] = act[wave];

    // stage layer-1 A tile: 16 rows x 32 K (9 real features, rest zero), f32->f16
    for (int s = lane; s < 16 * 32; s += 32) {
        int r = s >> 5, k = s & 31;
        float v = 0.f;
        int p = p0 + r;
        if (p < N_PTS && k < 9) v = pt[p * 9 + k];
        A[r][k] = (_Float16)v;
    }

    mlp_layer<1, 4 >(A, wf, 0,  bias, 0,   lane);   // 9(->32) -> 64
    mlp_layer<2, 8 >(A, wf, 4,  bias, 64,  lane);   // 64      -> 128
    mlp_layer<4, 16>(A, wf, 20, bias, 192, lane);   // 128     -> 256

    // layer 4: 256 -> 64, no ReLU, scatter atomic-max into [B,C,GX,GY] pooled grid
    {
        v16h a[8];
        load_a<8>(A, lane, a);
        int nn  = lane & 15;
        int hi8 = (lane & 16) ? 8 : 0;
        int vk[8];
#pragma unroll
        for (int r = 0; r < 8; ++r) {
            int p = p0 + r + hi8;
            vk[r] = (p < N_PTS) ? voxkeep[p] : -1;
        }
#pragma unroll
        for (int nt = 0; nt < 4; ++nt) {
            v8f acc = {};
#pragma unroll
            for (int kt = 0; kt < 8; ++kt) {
                v16h b = *(const v16h*)(wf + ((size_t)(84 + kt * 4 + nt) * 32 + lane) * 16);
                acc = __builtin_amdgcn_wmma_f32_16x16x32_f16(false, a[kt], false, b,
                                                             (short)0, acc, false, false);
            }
            float bv = bias[448 + nt * 16 + nn];
            int c = nt * 16 + nn;
#pragma unroll
            for (int r = 0; r < 8; ++r) {
                int v = vk[r];
                if (v >= 0) {
                    int b_  = v / (GX * GY);
                    int rem = v - b_ * (GX * GY);
                    int x   = rem / GY;
                    int y   = rem - x * GY;
                    size_t o = (((size_t)b_ * 64 + c) * GX + x) * GY + y;
                    atomicMaxFloat(&pooled[o], acc[r] + bv);
                }
            }
        }
    }
}

// ---------------- 3x3 stride-1 max pool (empty voxel -inf -> 0 before pooling) --------
__global__ void pool_kernel(const float* __restrict__ pooled, float* __restrict__ out) {
    size_t idx = (size_t)blockIdx.x * blockDim.x + threadIdx.x;
    if (idx >= POOLN) return;
    int y = (int)(idx % GY); size_t t = idx / GY;
    int x = (int)(t % GX);   t /= GX;
    int c = (int)(t % 64);
    int b = (int)(t / 64);
    const float* base = pooled + ((size_t)b * 64 + c) * GX * GY;
    float m = -3.4e38f;
#pragma unroll
    for (int dx = -1; dx <= 1; ++dx) {
        int xx = x + dx;
        if (xx < 0 || xx >= GX) continue;
#pragma unroll
        for (int dy = -1; dy <= 1; ++dy) {
            int yy = y + dy;
            if (yy < 0 || yy >= GY) continue;
            float vv = base[(size_t)xx * GY + yy];
            if (vv < -1e30f) vv = 0.f;   // empty voxel (-inf) -> 0, per reference
            m = fmaxf(m, vv);
        }
    }
    out[idx] = m;
}

extern "C" void kernel_launch(void* const* d_in, const int* in_sizes, int n_in,
                              void* d_out, int out_size, void* d_ws, size_t ws_size,
                              hipStream_t stream) {
    const float* pt  = (const float*)d_in[0];
    const int*   bi  = (const int*)d_in[1];
    const int*   gx  = (const int*)d_in[2];
    const int*   gy  = (const int*)d_in[3];
    const float* bn[16];
    for (int i = 0; i < 16; ++i) bn[i] = (const float*)d_in[4 + i];
    const float* w1  = (const float*)d_in[20]; const float* bb1 = (const float*)d_in[21];
    const float* w2  = (const float*)d_in[22]; const float* bb2 = (const float*)d_in[23];
    const float* w3  = (const float*)d_in[24]; const float* bb3 = (const float*)d_in[25];
    const float* w4  = (const float*)d_in[26]; const float* bb4 = (const float*)d_in[27];

    char* ws = (char*)d_ws;
    _Float16* wf     = (_Float16*)(ws + OFF_W);
    float*    bias   = (float*)(ws + OFF_BIAS);
    int*      voxk   = (int*)(ws + OFF_VOX);
    int*      cnt    = (int*)(ws + OFF_CNT);
    float*    pooled = (float*)(ws + OFF_POOL);

    prep_kernel<<<64, 256, 0, stream>>>(
        bn[0], bn[1], bn[2], bn[3], bn[4], bn[5], bn[6], bn[7],
        bn[8], bn[9], bn[10], bn[11], bn[12], bn[13], bn[14], bn[15],
        w1, bb1, w2, bb2, w3, bb3, w4, bb4, wf, bias);

    init_kernel<<<8192, 256, 0, stream>>>((unsigned*)pooled, cnt);

    rank_kernel<<<(N_PTS + 255) / 256, 256, 0, stream>>>(bi, gx, gy, cnt, voxk);

    int ntiles = (N_PTS + 15) / 16;                 // 18750
    int blocks = (ntiles + 3) / 4;                  // 4 waves per block
    mlp_kernel<<<blocks, 128, 0, stream>>>(pt, voxk, wf, bias, pooled, ntiles);

    size_t pool_blocks = (POOLN + 255) / 256;       // 172800
    pool_kernel<<<(int)pool_blocks, 256, 0, stream>>>(pooled, (float*)d_out);
}
